// AdaptiveModel_73340861546848
// MI455X (gfx1250) — matrix operand order repaired and checked by reading, not compiled, not run on hardware
//
#include <hip/hip_runtime.h>
#include <hip/hip_bf16.h>
#include <cmath>

#define B_   4
#define N_   256
#define D_   128
#define NL_  2
#define V_   32000
#define S_   4
#define ROWS (B_ * N_)   // 1024

typedef __attribute__((ext_vector_type(16))) _Float16 v16h;
typedef __attribute__((ext_vector_type(8)))  float    v8f;
typedef __attribute__((ext_vector_type(4)))  unsigned int u32x4;
typedef __attribute__((ext_vector_type(8)))  int      i32x8;
typedef __attribute__((ext_vector_type(4)))  int      i32x4;

__device__ __forceinline__ float gelu_f(float x) {
  // exact erf GELU (torch nn.GELU default)
  return 0.5f * x * (1.0f + erff(x * 0.70710678118654752440f));
}

// ---------------------------------------------------------------------------
// Tensor Data Mover: 2D f32 tile (tile_x contiguous elems  x  tile_y rows,
// row stride = stride_elems) from global -> LDS at byte offset lds_off.
// LDS padding: +1 DWORD every `tile_x` DWORDs so the deposited row stride is
// (tile_x + 1) floats (bank-conflict friendly). pad_interval code: 2^(c+1)
// DWORDs, so tile_x=32 -> c=4, tile_x=64 -> c=5.
// D# layout per CDNA5 ISA ch.8 (group0 128b, group1 256b, groups2/3 zero=2D).
// clang-23 lane: 6-arg builtin (u32x4, i32x8, i32x4, i32x4, i32x8, i32 cpol).
// ---------------------------------------------------------------------------
__device__ __forceinline__ void tdm_load_2d(unsigned int lds_off,
                                            const void* gptr,
                                            unsigned int tile_x,
                                            unsigned int tile_y,
                                            unsigned int stride_elems,
                                            unsigned int pad_code) {
  const unsigned long long ga = (unsigned long long)(uintptr_t)gptr;
  u32x4 g0;
  g0[0] = 1u;                                   // count=1 (valid user D#)
  g0[1] = lds_off;                              // lds_addr (bytes)
  g0[2] = (unsigned int)ga;                     // global_addr[31:0]
  g0[3] = (unsigned int)((ga >> 32) & 0x01FFFFFFu) | (2u << 30);  // addr[56:32] | type=2
  i32x8 g1;
  g1[0] = (int)((2u << 16)                      // data_size = 4B
                | (1u << 20)                    // pad_enable
                | (pad_code << 22));            // pad_interval; pad_amount=0 (1 DWORD)
  g1[1] = (int)((tile_x & 0xFFFFu) << 16);      // tensor_dim0[15:0] @ bits 63:48
  g1[2] = (int)(((tile_x >> 16) & 0xFFFFu) | ((tile_y & 0xFFFFu) << 16)); // dim0 hi | dim1 lo
  g1[3] = (int)(((tile_y >> 16) & 0xFFFFu) | ((tile_x & 0xFFFFu) << 16)); // dim1 hi | tile_dim0
  g1[4] = (int)(tile_y & 0xFFFFu);              // tile_dim1 | tile_dim2=0
  g1[5] = (int)stride_elems;                    // tensor_dim0_stride[31:0]
  g1[6] = 0;                                    // stride0[47:32] | stride1[15:0]
  g1[7] = 0;
  const i32x4 z4 = {0, 0, 0, 0};                // groups 2/3: NULL-equivalent (2D)
  const i32x8 z8 = {0, 0, 0, 0, 0, 0, 0, 0};    // extra group (clang-23 form): zeros
  __builtin_amdgcn_tensor_load_to_lds(g0, g1, z4, z4, z8, 0);
}

// ---------------------------------------------------------------------------
// x = embed[ids] + pos
// ---------------------------------------------------------------------------
__global__ void k_embed(const int* __restrict__ ids, const float* __restrict__ emb,
                        const float* __restrict__ pos, float* __restrict__ X) {
  const int row = blockIdx.x;          // b*N + n
  const int d   = threadIdx.x;         // 0..127
  const int n   = row % N_;
  const int id  = ids[row];
  X[row * D_ + d] = emb[id * D_ + d] + pos[n * D_ + d];
}

// ---------------------------------------------------------------------------
// Generic WMMA GEMM: C[M,N] = act(A[M,K] @ W[K,N] + bias)
// 256 threads = 8 waves; block tile 32(M) x 64(N); K chunked by 32.
// Double-buffered TDM pipeline: wave 0 issues stage k+1's DMA into the
// alternate LDS buffer, then waits TENSORcnt<=2 (TDM is in-order per wave, so
// stage k's two tiles are resident), computes, barrier. Tail waits <=0.
//
// WMMA fragment layouts (ISA 7.12.2, wave32):
//   A 16x32 f16 : lane L -> M=L%16, half=L/16; elem e -> K=((e>>3)<<4)+half*8+(e&7)
//   B 32x16 f16 : lane L -> N=L%16, half=L/16; elem e -> K=((e>>3)<<4)+half*8+(e&7)
//   D 16x16 f32 : lane L -> N=L%16; vgpr r -> M=r+8*(L/16)
// ---------------------------------------------------------------------------
template <int ACT>
__global__ void k_gemm(const float* __restrict__ A, const float* __restrict__ W,
                       const float* __restrict__ bias, float* __restrict__ C,
                       int M, int N, int K) {
  __shared__ float Asf[2][32 * 33];   // TDM-padded [row][33], double-buffered
  __shared__ float Bsf[2][32 * 65];   // TDM-padded [k][65],  double-buffered

  const int tid  = threadIdx.x;
  const int wv   = tid >> 5;
  const int lane = tid & 31;
  const int half = lane >> 4;
  const int lr   = lane & 15;
  const int mw   = wv & 1;           // 2 M-tiles
  const int nw   = wv >> 1;          // 4 N-tiles
  const int m0   = blockIdx.y * 32;
  const int n0   = blockIdx.x * 64;

  const unsigned int As_off[2] = {(unsigned int)(uintptr_t)&Asf[0][0],
                                  (unsigned int)(uintptr_t)&Asf[1][0]};
  const unsigned int Bs_off[2] = {(unsigned int)(uintptr_t)&Bsf[0][0],
                                  (unsigned int)(uintptr_t)&Bsf[1][0]};

  // prologue: stage 0 in flight
  if (tid < 32) {
    tdm_load_2d(As_off[0], &A[(size_t)m0 * K], 32, 32, (unsigned)K, 4);
    tdm_load_2d(Bs_off[0], &W[n0], 64, 32, (unsigned)N, 5);
  }

  v8f c = {};
  for (int k0 = 0; k0 < K; k0 += 32) {
    const int buf = (k0 >> 5) & 1;
    if (tid < 32) {  // wave 0 drives the DMA pipeline
      if (k0 + 32 < K) {
        tdm_load_2d(As_off[buf ^ 1], &A[(size_t)m0 * K + k0 + 32], 32, 32,
                    (unsigned)K, 4);
        tdm_load_2d(Bs_off[buf ^ 1], &W[(size_t)(k0 + 32) * N + n0], 64, 32,
                    (unsigned)N, 5);
        __builtin_amdgcn_s_wait_tensorcnt(2);   // stage k resident, k+1 in flight
      } else {
        __builtin_amdgcn_s_wait_tensorcnt(0);   // tail: drain
      }
    }
    __syncthreads();

    v16h af, bf;
#pragma unroll
    for (int e = 0; e < 16; ++e) {
      const int ka = ((e >> 3) << 4) + half * 8 + (e & 7);
      af[e] = (_Float16)Asf[buf][(mw * 16 + lr) * 33 + ka];
      const int kb = ((e >> 3) << 4) + half * 8 + (e & 7);
      bf[e] = (_Float16)Bsf[buf][kb * 65 + nw * 16 + lr];
    }
    c = __builtin_amdgcn_wmma_f32_16x16x32_f16(false, af, false, bf, (short)0, c,
                                               false, false);
    __syncthreads();   // fence buffer before it is re-targeted at stage k+2
  }

  const int col = n0 + nw * 16 + lr;
  const float bb = bias ? bias[col] : 0.0f;
#pragma unroll
  for (int r = 0; r < 8; ++r) {
    const int rowm = m0 + mw * 16 + r + 8 * half;
    float v = c[r] + bb;
    if (ACT == 1) v = gelu_f(v);
    C[(size_t)rowm * N + col] = v;
  }
}

// ---------------------------------------------------------------------------
// Fused pair module (flash-style online softmax over j):
//   h_ij = gelu(Apair[i] + Cpair[j])   (Apair already holds x@w1_top + b1,
//                                       Cpair holds x@w1_bot)
//   p_ij = h_ij @ w2 + b2 ;  nrm_ij = ||p_ij|| (0 if sq==0) ; mask j>i
//   P[i] = softmax_j(nrm) . p
// One workgroup per (b,i); 8 waves each own a 16-wide slab of D.
// ---------------------------------------------------------------------------
__global__ void k_pair(const float* __restrict__ Ap, const float* __restrict__ Cp,
                       const float* __restrict__ w2, const float* __restrict__ b2,
                       float* __restrict__ P) {
  __shared__ _Float16 w2h[128 * 130];
  __shared__ _Float16 hsh[16 * 130];
  __shared__ float a_sh[128];
  __shared__ float sq_sh[128];   // [wave*16 + M]
  __shared__ float nrm_sh[16];
  __shared__ float e_sh[16];
  __shared__ float m_sh, s_sh, scale_sh;

  const int tid  = threadIdx.x;
  const int wv   = tid >> 5;
  const int lane = tid & 31;
  const int half = lane >> 4;
  const int lr   = lane & 15;
  const int rowi = blockIdx.x;
  const int b    = rowi / N_;
  const int i    = rowi % N_;

  // stage w2 (128x128) as f16
  for (int q = tid; q < 128 * 32; q += 256) {
    const int r  = q >> 5;
    const int c4 = (q & 31) << 2;
    const float4 v = *reinterpret_cast<const float4*>(&w2[r * 128 + c4]);
    w2h[r * 130 + c4 + 0] = (_Float16)v.x;
    w2h[r * 130 + c4 + 1] = (_Float16)v.y;
    w2h[r * 130 + c4 + 2] = (_Float16)v.z;
    w2h[r * 130 + c4 + 3] = (_Float16)v.w;
  }
  if (tid < 128) a_sh[tid] = Ap[rowi * D_ + tid];
  if (tid == 0) { m_sh = -1e30f; s_sh = 0.0f; }
  __syncthreads();

  // hoist B fragments (this wave's 16 columns of w2) for the whole j-loop
  v16h bf[4];
#pragma unroll
  for (int kc = 0; kc < 4; ++kc)
#pragma unroll
    for (int e = 0; e < 16; ++e)
      bf[kc][e] = w2h[(kc * 32 + ((e >> 3) << 4) + half * 8 + (e & 7)) * 130 + wv * 16 + lr];

  const float bcol = b2[wv * 16 + lr];
  v8f acc = {};
  const int ntiles = (i >> 4) + 1;
  const int hj = tid >> 4;          // j_local 0..15
  const int hd = (tid & 15) << 3;   // d base, 8 per thread

  for (int jt = 0; jt < ntiles; ++jt) {
    {  // build h tile: gelu(a_i + c_j)
      const int j = jt * 16 + hj;
      const float* cp = &Cp[(size_t)(b * N_ + j) * D_ + hd];
      if (jt + 1 < ntiles)
        __builtin_prefetch(cp + 16 * D_, 0, 1);   // next j-tile -> global_prefetch
      const float4 v0 = *reinterpret_cast<const float4*>(cp);
      const float4 v1 = *reinterpret_cast<const float4*>(cp + 4);
      _Float16* hp = &hsh[hj * 130 + hd];
      hp[0] = (_Float16)gelu_f(a_sh[hd + 0] + v0.x);
      hp[1] = (_Float16)gelu_f(a_sh[hd + 1] + v0.y);
      hp[2] = (_Float16)gelu_f(a_sh[hd + 2] + v0.z);
      hp[3] = (_Float16)gelu_f(a_sh[hd + 3] + v0.w);
      hp[4] = (_Float16)gelu_f(a_sh[hd + 4] + v1.x);
      hp[5] = (_Float16)gelu_f(a_sh[hd + 5] + v1.y);
      hp[6] = (_Float16)gelu_f(a_sh[hd + 6] + v1.z);
      hp[7] = (_Float16)gelu_f(a_sh[hd + 7] + v1.w);
    }
    __syncthreads();

    // p-tile = h @ w2 (16x128 @ 128x16 per wave), K in 4 chunks of 32
    v8f c = {};
#pragma unroll
    for (int kc = 0; kc < 4; ++kc) {
      v16h af;
#pragma unroll
      for (int e = 0; e < 16; ++e) {
        const int ka = kc * 32 + ((e >> 3) << 4) + half * 8 + (e & 7);
        af[e] = hsh[lr * 130 + ka];
      }
      c = __builtin_amdgcn_wmma_f32_16x16x32_f16(false, af, false, bf[kc],
                                                 (short)0, c, false, false);
    }

    // per-j squared norm: batched butterfly over this wave's 16 columns
    float pv[8], v[8];
#pragma unroll
    for (int r = 0; r < 8; ++r) { pv[r] = c[r] + bcol; v[r] = pv[r] * pv[r]; }
#pragma unroll
    for (int mask = 1; mask <= 8; mask <<= 1) {
      float t[8];
#pragma unroll
      for (int r = 0; r < 8; ++r) t[r] = __shfl_xor(v[r], mask);
#pragma unroll
      for (int r = 0; r < 8; ++r) v[r] += t[r];
    }
    if (lr == 0) {
      float* sp = &sq_sh[wv * 16 + half * 8];
      *reinterpret_cast<float4*>(sp)     = float4{v[0], v[1], v[2], v[3]};
      *reinterpret_cast<float4*>(sp + 4) = float4{v[4], v[5], v[6], v[7]};
    }
    __syncthreads();

    if (tid < 16) {
      float tot = 0.0f;
#pragma unroll
      for (int w = 0; w < 8; ++w) tot += sq_sh[w * 16 + tid];
      const int j = jt * 16 + tid;
      nrm_sh[tid] = (j > i) ? -1e9f : (tot > 0.0f ? sqrtf(tot) : 0.0f);
    }
    __syncthreads();

    if (tid == 0) {  // online softmax state update
      float tmax = -1e30f;
      for (int q = 0; q < 16; ++q) tmax = fmaxf(tmax, nrm_sh[q]);
      const float mold = m_sh;
      const float mnew = fmaxf(mold, tmax);
      const float scl  = expf(mold - mnew);
      float esum = 0.0f;
      for (int q = 0; q < 16; ++q) {
        const float e = expf(nrm_sh[q] - mnew);
        e_sh[q] = e;
        esum += e;
      }
      s_sh = s_sh * scl + esum;
      m_sh = mnew;
      scale_sh = scl;
    }
    __syncthreads();

    const float scl = scale_sh;
#pragma unroll
    for (int r = 0; r < 8; ++r)
      acc[r] = acc[r] * scl + e_sh[r + 8 * half] * pv[r];
  }

  // reduce acc over j (8 regs) and the two lane halves, divide by softmax denom
  float tot = acc[0] + acc[1] + acc[2] + acc[3] + acc[4] + acc[5] + acc[6] + acc[7];
  tot += __shfl_xor(tot, 16);
  if (half == 0) P[(size_t)rowi * D_ + wv * 16 + lr] = tot / s_sh;
}

// ---------------------------------------------------------------------------
// triple: G[b,n,:] = mean_s gelu(XA[b,n] + XB[b,ji] + XC[b,ki] + b1)
// ---------------------------------------------------------------------------
__global__ void k_triple(const float* __restrict__ XA, const float* __restrict__ XB,
                         const float* __restrict__ XC, const int* __restrict__ ji,
                         const int* __restrict__ ki, const float* __restrict__ b1,
                         float* __restrict__ G) {
  const int row = blockIdx.x;
  const int d   = threadIdx.x;
  const int b   = row / N_;
  const float base = XA[row * D_ + d] + b1[d];
  float acc = 0.0f;
#pragma unroll
  for (int s = 0; s < S_; ++s) {
    const int j = ji[row * S_ + s];
    const int k = ki[row * S_ + s];
    acc += gelu_f(base + XB[(b * N_ + j) * D_ + d] + XC[(b * N_ + k) * D_ + d]);
  }
  G[row * D_ + d] = acc * (1.0f / S_);
}

// ---------------------------------------------------------------------------
// block reduction helper (128 threads)
// ---------------------------------------------------------------------------
__device__ __forceinline__ float blk_sum(float v, float* red) {
  const int t = threadIdx.x;
  red[t] = v;
  __syncthreads();
#pragma unroll
  for (int s = 64; s > 0; s >>= 1) {
    if (t < s) red[t] += red[t + s];
    __syncthreads();
  }
  const float r = red[0];
  __syncthreads();
  return r;
}

// x = LN(x + alpha*T + (1-alpha)*P), alpha = sigmoid(S1 . sw2 + sb2)
__global__ void k_mix_ln(const float* __restrict__ X, const float* __restrict__ T,
                         const float* __restrict__ P, const float* __restrict__ S1,
                         const float* __restrict__ sw2, const float* __restrict__ sb2,
                         const float* __restrict__ g, const float* __restrict__ bb,
                         float* __restrict__ Xo) {
  __shared__ float red[128];
  const int row = blockIdx.x;
  const int t   = threadIdx.x;
  const float dv  = (t < 64) ? S1[row * 64 + t] * sw2[t] : 0.0f;
  const float dot = blk_sum(dv, red);
  const float alpha = 1.0f / (1.0f + expf(-(dot + sb2[0])));
  const float y = X[row * D_ + t] + alpha * T[row * D_ + t] +
                  (1.0f - alpha) * P[row * D_ + t];
  const float mean = blk_sum(y, red) * (1.0f / D_);
  const float d    = y - mean;
  const float var  = blk_sum(d * d, red) * (1.0f / D_);
  Xo[row * D_ + t] = d * rsqrtf(var + 1e-5f) * g[t] + bb[t];
}

// x = LN(x + H)
__global__ void k_add_ln(const float* __restrict__ X, const float* __restrict__ H,
                         const float* __restrict__ g, const float* __restrict__ bb,
                         float* __restrict__ Xo) {
  __shared__ float red[128];
  const int row = blockIdx.x;
  const int t   = threadIdx.x;
  const float y = X[row * D_ + t] + H[row * D_ + t];
  const float mean = blk_sum(y, red) * (1.0f / D_);
  const float d    = y - mean;
  const float var  = blk_sum(d * d, red) * (1.0f / D_);
  Xo[row * D_ + t] = d * rsqrtf(var + 1e-5f) * g[t] + bb[t];
}

// ---------------------------------------------------------------------------
extern "C" void kernel_launch(void* const* d_in, const int* in_sizes, int n_in,
                              void* d_out, int out_size, void* d_ws, size_t ws_size,
                              hipStream_t stream) {
  (void)in_sizes; (void)n_in; (void)out_size; (void)ws_size;
  const int*   ids = (const int*)d_in[0];
  const int*   ji  = (const int*)d_in[1];
  const int*   ki  = (const int*)d_in[2];
  const float* emb = (const float*)d_in[3];
  const float* pos = (const float*)d_in[4];
  const float* pw1 = (const float*)d_in[5];
  const float* pb1 = (const float*)d_in[6];
  const float* pw2 = (const float*)d_in[7];
  const float* pb2 = (const float*)d_in[8];
  const float* tw1 = (const float*)d_in[9];
  const float* tb1 = (const float*)d_in[10];
  const float* tw2 = (const float*)d_in[11];
  const float* tb2 = (const float*)d_in[12];
  const float* sw1 = (const float*)d_in[13];
  const float* sb1 = (const float*)d_in[14];
  const float* sw2 = (const float*)d_in[15];
  const float* sb2 = (const float*)d_in[16];
  const float* n1g = (const float*)d_in[17];
  const float* n1b = (const float*)d_in[18];
  const float* fw1 = (const float*)d_in[19];
  const float* fb1 = (const float*)d_in[20];
  const float* fw2 = (const float*)d_in[21];
  const float* fb2 = (const float*)d_in[22];
  const float* n2g = (const float*)d_in[23];
  const float* n2b = (const float*)d_in[24];
  const float* hw  = (const float*)d_in[25];
  const float* hb  = (const float*)d_in[26];

  float* ws = (float*)d_ws;
  const size_t R = (size_t)ROWS * D_;   // 131072
  float* X  = ws;
  float* Ap = ws + 1 * R;
  float* Cq = ws + 2 * R;
  float* Pp = ws + 3 * R;
  float* XA = ws + 4 * R;
  float* XB = ws + 5 * R;
  float* XC = ws + 6 * R;
  float* G  = ws + 7 * R;
  float* Tt = ws + 8 * R;
  float* S1 = ws + 9 * R;               // ROWS*64
  float* H1 = ws + 9 * R + (size_t)ROWS * 64;   // ROWS*512
  float* H2 = H1 + (size_t)ROWS * 4 * D_;

  const dim3 gD(2, 32);     // N=128 outputs
  const dim3 gS(1, 32);     // N=64
  const dim3 gF(8, 32);     // N=512
  const dim3 gH(V_ / 64, 32);

  k_embed<<<ROWS, 128, 0, stream>>>(ids, emb, pos, X);

  for (int l = 0; l < NL_; ++l) {
    const float* pw1l = pw1 + (size_t)l * 2 * D_ * D_;
    const float* tw1l = tw1 + (size_t)l * 3 * D_ * D_;

    // pair projections (split concat): Ap = X@w1_top + b1 ; Cq = X@w1_bot
    k_gemm<0><<<gD, 256, 0, stream>>>(X, pw1l, pb1 + l * D_, Ap, ROWS, D_, D_);
    k_gemm<0><<<gD, 256, 0, stream>>>(X, pw1l + D_ * D_, nullptr, Cq, ROWS, D_, D_);
    k_pair<<<ROWS, 256, 0, stream>>>(Ap, Cq, pw2 + (size_t)l * D_ * D_,
                                     pb2 + l * D_, Pp);

    // triple projections (split concat), gather+mean, then @w2
    k_gemm<0><<<gD, 256, 0, stream>>>(X, tw1l, nullptr, XA, ROWS, D_, D_);
    k_gemm<0><<<gD, 256, 0, stream>>>(X, tw1l + D_ * D_, nullptr, XB, ROWS, D_, D_);
    k_gemm<0><<<gD, 256, 0, stream>>>(X, tw1l + 2 * D_ * D_, nullptr, XC, ROWS, D_, D_);
    k_triple<<<ROWS, 128, 0, stream>>>(XA, XB, XC, ji + (size_t)l * ROWS * S_,
                                       ki + (size_t)l * ROWS * S_, tb1 + l * D_, G);
    k_gemm<0><<<gD, 256, 0, stream>>>(G, tw2 + (size_t)l * D_ * D_, tb2 + l * D_,
                                      Tt, ROWS, D_, D_);

    // selection gate + mix + LN1 (in place on X)
    k_gemm<1><<<gS, 256, 0, stream>>>(X, sw1 + (size_t)l * D_ * 64, sb1 + l * 64,
                                      S1, ROWS, 64, D_);
    k_mix_ln<<<ROWS, 128, 0, stream>>>(X, Tt, Pp, S1, sw2 + l * 64, sb2 + l,
                                       n1g + l * D_, n1b + l * D_, X);

    // FFN + LN2 (in place on X)
    k_gemm<1><<<gF, 256, 0, stream>>>(X, fw1 + (size_t)l * D_ * 4 * D_,
                                      fb1 + l * 4 * D_, H1, ROWS, 4 * D_, D_);
    k_gemm<0><<<gD, 256, 0, stream>>>(H1, fw2 + (size_t)l * 4 * D_ * D_,
                                      fb2 + l * D_, H2, ROWS, D_, 4 * D_);
    k_add_ln<<<ROWS, 128, 0, stream>>>(X, H2, n2g + l * D_, n2b + l * D_, X);
  }

  // head: [1024,128] @ [128,32000] + bias -> d_out
  k_gemm<0><<<gH, 256, 0, stream>>>(X, hw, hb, (float*)d_out, ROWS, V_, D_);
}
